// SwinTransformerBlock_27977416966347
// MI455X (gfx1250) — compile-verified, matrix operand order checked by reading, not run on hardware
//
#include <hip/hip_runtime.h>
#include <hip/hip_bf16.h>
#include <stdint.h>

// ---------------- problem constants ----------------
#define Bsz   32
#define Himg  56
#define Wimg  56
#define Cdim  384
#define NHd   12
#define WSz   7
#define SSz   3
#define NTOKW 49            // tokens per window
#define DhD   32
#define MLPH  1536
#define NWIN  2048          // Bsz * 64 windows
#define TOK   100352        // Bsz*Himg*Wimg
#define MT64  1568          // TOK / 64

// ---------------- WMMA fragment types ----------------
typedef __bf16 bfrag8  __attribute__((ext_vector_type(8)));
typedef __bf16 bfrag16 __attribute__((ext_vector_type(16)));
typedef float  floatx8 __attribute__((ext_vector_type(8)));

union FragU { bfrag16 v; bfrag8 h[2]; unsigned short s[16]; };

__device__ __forceinline__ unsigned short f2bf(float f) {
  unsigned int u = __float_as_uint(f);
  u += 0x7FFFu + ((u >> 16) & 1u);           // round-to-nearest-even
  return (unsigned short)(u >> 16);
}

// Load one 16(row)x32(k) bf16 fragment. ISA layout: lane = row%16 + 16*g,
// per-lane k-chunks at g*8 .. g*8+7 and g*8+16 .. g*8+23 (each 16B).
__device__ __forceinline__ bfrag16 load_frag(const __bf16* base, int ld) {
  int lane = threadIdx.x & 31;
  int n = lane & 15, g = lane >> 4;
  const __bf16* p = base + (size_t)n * ld + g * 8;
  FragU u;
  u.h[0] = *(const bfrag8*)(p);
  u.h[1] = *(const bfrag8*)(p + 16);
  return u.v;
}

__device__ __forceinline__ bfrag16 load_frag_guard(const __bf16* base, int ld,
                                                   int row0, int rowmax) {
  int lane = threadIdx.x & 31;
  int n = lane & 15, g = lane >> 4;
  FragU u;
  if (row0 + n < rowmax) {
    const __bf16* p = base + (size_t)n * ld + g * 8;
    u.h[0] = *(const bfrag8*)(p);
    u.h[1] = *(const bfrag8*)(p + 16);
  } else {
#pragma unroll
    for (int i = 0; i < 16; ++i) u.s[i] = 0;
  }
  return u.v;
}

__device__ __forceinline__ floatx8 wmma_bf16(bfrag16 a, bfrag16 b, floatx8 c) {
  return __builtin_amdgcn_wmma_f32_16x16x32_bf16(false, a, false, b,
                                                 (short)0, c, false, false);
}

// ---- 64(M) x 32(N) wave tile (used by attention; head dim is 32) ----
__device__ __forceinline__ void gemm_core(const __bf16* A, int lda, int rowmax,
                                          const __bf16* Wt, int ldw, int ksteps,
                                          floatx8 acc[4][2]) {
#pragma unroll
  for (int mt = 0; mt < 4; ++mt)
#pragma unroll
    for (int nt = 0; nt < 2; ++nt)
#pragma unroll
      for (int e = 0; e < 8; ++e) acc[mt][nt][e] = 0.0f;
  for (int ks = 0; ks < ksteps; ++ks) {
    bfrag16 a[4], b[2];
#pragma unroll
    for (int mt = 0; mt < 4; ++mt)
      a[mt] = load_frag_guard(A + (size_t)mt * 16 * lda + ks * 32, lda, mt * 16, rowmax);
#pragma unroll
    for (int nt = 0; nt < 2; ++nt)
      b[nt] = load_frag(Wt + (size_t)nt * 16 * ldw + ks * 32, ldw);
#pragma unroll
    for (int mt = 0; mt < 4; ++mt)
#pragma unroll
      for (int nt = 0; nt < 2; ++nt)
        acc[mt][nt] = wmma_bf16(a[mt], b[nt], acc[mt][nt]);
  }
}

// ---- 64(M) x 64(N) wave tile (proj / fc1 / fc2): 2x arithmetic intensity ----
__device__ __forceinline__ void gemm_core64(const __bf16* A, int lda,
                                            const __bf16* Wt, int ldw, int ksteps,
                                            floatx8 acc[4][4]) {
#pragma unroll
  for (int mt = 0; mt < 4; ++mt)
#pragma unroll
    for (int nt = 0; nt < 4; ++nt)
#pragma unroll
      for (int e = 0; e < 8; ++e) acc[mt][nt][e] = 0.0f;
  for (int ks = 0; ks < ksteps; ++ks) {
    bfrag16 a[4];
#pragma unroll
    for (int mt = 0; mt < 4; ++mt)
      a[mt] = load_frag(A + (size_t)mt * 16 * lda + ks * 32, lda);
#pragma unroll
    for (int nt = 0; nt < 4; ++nt) {
      bfrag16 b = load_frag(Wt + (size_t)nt * 16 * ldw + ks * 32, ldw);
#pragma unroll
      for (int mt = 0; mt < 4; ++mt)
        acc[mt][nt] = wmma_bf16(a[mt], b, acc[mt][nt]);
    }
  }
}

// ---------------- weight pack: f32 (K,N) -> bf16 (N,K) ----------------
__global__ void pack_wT_kernel(const float* __restrict__ w,
                               unsigned short* __restrict__ wT, int K, int Ncols) {
  int idx = blockIdx.x * 256 + threadIdx.x;
  if (idx >= K * Ncols) return;
  int n = idx / K, k = idx % K;
  wT[idx] = f2bf(w[(size_t)k * Ncols + n]);
}

// ---------------- LayerNorm (+ optional shift/window gather) ----------------
__global__ __launch_bounds__(256) void ln_kernel(const float* __restrict__ src,
    const float* __restrict__ scale, const float* __restrict__ bias,
    unsigned short* __restrict__ dst, int windowed) {
  int wave = threadIdx.x >> 5, lane = threadIdx.x & 31;
  int dtok = blockIdx.x * 8 + wave;
  if (dtok >= TOK) return;
  int srow;
  if (windowed) {
    int w = dtok / NTOKW, i = dtok % NTOKW;
    int b = w >> 6, wi = w & 63, wh = wi >> 3, ww = wi & 7;
    int r = i / 7, c = i % 7;
    int hh = wh * 7 + r + SSz; if (hh >= Himg) hh -= Himg;   // roll(-3) gather
    int wc = ww * 7 + c + SSz; if (wc >= Wimg) wc -= Wimg;
    srow = b * (Himg * Wimg) + hh * Wimg + wc;
  } else {
    srow = dtok;
  }
  const float* p = src + (size_t)srow * Cdim;
  float v[12];
  float s = 0.f;
#pragma unroll
  for (int k = 0; k < 12; ++k) { v[k] = p[lane + 32 * k]; s += v[k]; }
#pragma unroll
  for (int off = 16; off; off >>= 1) s += __shfl_xor(s, off, 32);
  float mu = s * (1.0f / Cdim);
  float q = 0.f;
#pragma unroll
  for (int k = 0; k < 12; ++k) { float d = v[k] - mu; q += d * d; }
#pragma unroll
  for (int off = 16; off; off >>= 1) q += __shfl_xor(q, off, 32);
  float rs = rsqrtf(q * (1.0f / Cdim) + 1e-6f);
  unsigned short* o = dst + (size_t)dtok * Cdim;
#pragma unroll
  for (int k = 0; k < 12; ++k) {
    int c = lane + 32 * k;
    o[c] = f2bf((v[k] - mu) * rs * scale[c] + bias[c]);
  }
}

// ---------------- attention: one wave per (window, head) ----------------
__device__ __forceinline__ int win_label(int wh, int ww, int t) {
  int r = t / 7, c = t % 7;
  int hh = wh * 7 + r, wc = ww * 7 + c;
  int lh = (hh < Himg - WSz) ? 0 : ((hh < Himg - SSz) ? 1 : 2);
  int lw = (wc < Wimg - WSz) ? 0 : ((wc < Wimg - SSz) ? 1 : 2);
  return lh * 3 + lw;
}

__global__ __launch_bounds__(32) void attn_kernel(const unsigned short* __restrict__ act,
    const unsigned short* __restrict__ qkvT, const float* __restrict__ qkv_b,
    unsigned short* __restrict__ outB) {
  int w = blockIdx.x / NHd, hd = blockIdx.x % NHd;
  int wi = w & 63, wh = wi >> 3, ww = wi & 7;
  __shared__ __align__(16) unsigned short qS[64 * 32];
  __shared__ __align__(16) unsigned short kS[64 * 32];
  __shared__ __align__(16) unsigned short vTS[32 * 64];
  __shared__ __align__(16) unsigned short pS[64 * 64];
  __shared__ __align__(16) float sS[64 * 64];
  int lane = threadIdx.x, nloc = lane & 15, g = lane >> 4;

  for (int i = lane; i < 64 * 32; i += 32) { qS[i] = 0; kS[i] = 0; vTS[i] = 0; }
  __syncthreads();

  const __bf16* A  = (const __bf16*)act + (size_t)w * NTOKW * Cdim;
  const __bf16* Wt = (const __bf16*)qkvT;
  const float sc = 0.17677669529663689f;  // Dh^-0.5, folded into Q
  floatx8 acc[4][2];

  // ---- Q ----
  gemm_core(A, Cdim, NTOKW, Wt + (size_t)(hd * 32) * Cdim, Cdim, 12, acc);
#pragma unroll
  for (int mt = 0; mt < 4; ++mt)
#pragma unroll
    for (int nt = 0; nt < 2; ++nt) {
      float bj = qkv_b[hd * 32 + nt * 16 + nloc];
#pragma unroll
      for (int r = 0; r < 8; ++r) {
        int m = mt * 16 + g * 8 + r;
        if (m < NTOKW) qS[m * 32 + nt * 16 + nloc] = f2bf((acc[mt][nt][r] + bj) * sc);
      }
    }
  // ---- K ----
  gemm_core(A, Cdim, NTOKW, Wt + (size_t)(Cdim + hd * 32) * Cdim, Cdim, 12, acc);
#pragma unroll
  for (int mt = 0; mt < 4; ++mt)
#pragma unroll
    for (int nt = 0; nt < 2; ++nt) {
      float bj = qkv_b[Cdim + hd * 32 + nt * 16 + nloc];
#pragma unroll
      for (int r = 0; r < 8; ++r) {
        int m = mt * 16 + g * 8 + r;
        if (m < NTOKW) kS[m * 32 + nt * 16 + nloc] = f2bf(acc[mt][nt][r] + bj);
      }
    }
  // ---- V (stored transposed: vTS[d][token]) ----
  gemm_core(A, Cdim, NTOKW, Wt + (size_t)(2 * Cdim + hd * 32) * Cdim, Cdim, 12, acc);
#pragma unroll
  for (int mt = 0; mt < 4; ++mt)
#pragma unroll
    for (int nt = 0; nt < 2; ++nt) {
      float bj = qkv_b[2 * Cdim + hd * 32 + nt * 16 + nloc];
#pragma unroll
      for (int r = 0; r < 8; ++r) {
        int m = mt * 16 + g * 8 + r;
        if (m < NTOKW) vTS[(nt * 16 + nloc) * 64 + m] = f2bf(acc[mt][nt][r] + bj);
      }
    }
  __syncthreads();

  // ---- S = Q K^T (+ shifted-window mask) ----
  floatx8 s[4][4];
#pragma unroll
  for (int mt = 0; mt < 4; ++mt)
#pragma unroll
    for (int nt = 0; nt < 4; ++nt)
#pragma unroll
      for (int e = 0; e < 8; ++e) s[mt][nt][e] = 0.0f;
  bfrag16 qa[4];
#pragma unroll
  for (int mt = 0; mt < 4; ++mt)
    qa[mt] = load_frag((const __bf16*)qS + mt * 16 * 32, 32);
#pragma unroll
  for (int nt = 0; nt < 4; ++nt) {
    bfrag16 kb = load_frag((const __bf16*)kS + nt * 16 * 32, 32);
#pragma unroll
    for (int mt = 0; mt < 4; ++mt) s[mt][nt] = wmma_bf16(qa[mt], kb, s[mt][nt]);
  }
#pragma unroll
  for (int mt = 0; mt < 4; ++mt)
#pragma unroll
    for (int nt = 0; nt < 4; ++nt) {
      int j = nt * 16 + nloc;
#pragma unroll
      for (int r = 0; r < 8; ++r) {
        int i = mt * 16 + g * 8 + r;
        float v = s[mt][nt][r];
        if (i < NTOKW && j < NTOKW) {
          if (win_label(wh, ww, i) != win_label(wh, ww, j)) v -= 100.0f;
        } else {
          v = -1e30f;
        }
        sS[i * 64 + j] = v;
      }
    }
  __syncthreads();

  // ---- softmax rows ----
  for (int i = lane; i < 64; i += 32) {
    if (i < NTOKW) {
      float mx = -1e30f;
      for (int j = 0; j < NTOKW; ++j) mx = fmaxf(mx, sS[i * 64 + j]);
      float sum = 0.f;
      for (int j = 0; j < NTOKW; ++j) sum += __expf(sS[i * 64 + j] - mx);
      float inv = 1.0f / sum;
      for (int j = 0; j < NTOKW; ++j) pS[i * 64 + j] = f2bf(__expf(sS[i * 64 + j] - mx) * inv);
      for (int j = NTOKW; j < 64; ++j) pS[i * 64 + j] = 0;
    } else {
      for (int j = 0; j < 64; ++j) pS[i * 64 + j] = 0;
    }
  }
  __syncthreads();

  // ---- O = P V ----
  floatx8 o[4][2];
#pragma unroll
  for (int mt = 0; mt < 4; ++mt)
#pragma unroll
    for (int nt = 0; nt < 2; ++nt)
#pragma unroll
      for (int e = 0; e < 8; ++e) o[mt][nt][e] = 0.0f;
#pragma unroll
  for (int kt = 0; kt < 2; ++kt) {
    bfrag16 pa[4];
#pragma unroll
    for (int mt = 0; mt < 4; ++mt)
      pa[mt] = load_frag((const __bf16*)pS + mt * 16 * 64 + kt * 32, 64);
#pragma unroll
    for (int nt = 0; nt < 2; ++nt) {
      bfrag16 vb = load_frag((const __bf16*)vTS + nt * 16 * 64 + kt * 32, 64);
#pragma unroll
      for (int mt = 0; mt < 4; ++mt) o[mt][nt] = wmma_bf16(pa[mt], vb, o[mt][nt]);
    }
  }
#pragma unroll
  for (int mt = 0; mt < 4; ++mt)
#pragma unroll
    for (int nt = 0; nt < 2; ++nt)
#pragma unroll
      for (int r = 0; r < 8; ++r) {
        int i = mt * 16 + g * 8 + r;
        if (i < NTOKW)
          outB[((size_t)w * NTOKW + i) * Cdim + hd * 32 + nt * 16 + nloc] =
              f2bf(o[mt][nt][r]);
      }
}

// ---------------- proj GEMM + window-reverse + roll(+3) + residual ----------------
__global__ __launch_bounds__(32) void proj_kernel(const unsigned short* __restrict__ act,
    const unsigned short* __restrict__ wT, const float* __restrict__ bias,
    const float* __restrict__ resid, float* __restrict__ x1) {
  int n64 = blockIdx.x % 6, mtile = blockIdx.x / 6;
  int lane = threadIdx.x, nloc = lane & 15, g = lane >> 4;
  floatx8 acc[4][4];
  gemm_core64((const __bf16*)act + (size_t)mtile * 64 * Cdim, Cdim,
              (const __bf16*)wT + (size_t)(n64 * 64) * Cdim, Cdim, 12, acc);
#pragma unroll
  for (int mt = 0; mt < 4; ++mt)
#pragma unroll
    for (int nt = 0; nt < 4; ++nt) {
      int j = n64 * 64 + nt * 16 + nloc;
      float bj = bias[j];
#pragma unroll
      for (int r = 0; r < 8; ++r) {
        int row = mtile * 64 + mt * 16 + g * 8 + r;     // windowed token id
        int w = row / NTOKW, i = row % NTOKW;
        int b = w >> 6, wiw = w & 63, wh = wiw >> 3, ww = wiw & 7;
        int rr = i / 7, cc = i % 7;
        int hh = wh * 7 + rr + SSz; if (hh >= Himg) hh -= Himg;  // roll(+3)
        int wc = ww * 7 + cc + SSz; if (wc >= Wimg) wc -= Wimg;
        size_t dst = ((size_t)(b * (Himg * Wimg) + hh * Wimg + wc)) * Cdim + j;
        x1[dst] = resid[dst] + acc[mt][nt][r] + bj;
      }
    }
}

// ---------------- fc1 chunk (384-wide slice of MLPH) + GELU(tanh) ----------------
__global__ __launch_bounds__(32) void fc1_kernel(const unsigned short* __restrict__ act,
    const unsigned short* __restrict__ w1T, const float* __restrict__ b1,
    unsigned short* __restrict__ hbuf, int ch) {
  int n64 = blockIdx.x % 6, mtile = blockIdx.x / 6;
  int lane = threadIdx.x, nloc = lane & 15, g = lane >> 4;
  floatx8 acc[4][4];
  gemm_core64((const __bf16*)act + (size_t)mtile * 64 * Cdim, Cdim,
              (const __bf16*)w1T + (size_t)(ch * 384 + n64 * 64) * Cdim, Cdim, 12, acc);
#pragma unroll
  for (int mt = 0; mt < 4; ++mt)
#pragma unroll
    for (int nt = 0; nt < 4; ++nt) {
      int j = n64 * 64 + nt * 16 + nloc;
      float bj = b1[ch * 384 + j];
#pragma unroll
      for (int r = 0; r < 8; ++r) {
        float x = acc[mt][nt][r] + bj;
        float gv = 0.5f * x *
                   (1.0f + tanhf(0.7978845608028654f * (x + 0.044715f * x * x * x)));
        int row = mtile * 64 + mt * 16 + g * 8 + r;
        hbuf[(size_t)row * 384 + j] = f2bf(gv);
      }
    }
}

// ---------------- fc2 chunk: ch0 seeds out = x1 + bias + acc, later chunks RMW ----------------
__global__ __launch_bounds__(32) void fc2_kernel(const unsigned short* __restrict__ hbuf,
    const unsigned short* __restrict__ w2T, const float* __restrict__ b2,
    const float* __restrict__ x1, float* __restrict__ out, int ch) {
  int n64 = blockIdx.x % 6, mtile = blockIdx.x / 6;
  int lane = threadIdx.x, nloc = lane & 15, g = lane >> 4;
  floatx8 acc[4][4];
  gemm_core64((const __bf16*)hbuf + (size_t)mtile * 64 * 384, 384,
              (const __bf16*)w2T + (size_t)(n64 * 64) * MLPH + ch * 384, MLPH, 12, acc);
#pragma unroll
  for (int mt = 0; mt < 4; ++mt)
#pragma unroll
    for (int nt = 0; nt < 4; ++nt) {
      int j = n64 * 64 + nt * 16 + nloc;
      float bj = b2[j];
#pragma unroll
      for (int r = 0; r < 8; ++r) {
        int row = mtile * 64 + mt * 16 + g * 8 + r;
        size_t idx = (size_t)row * Cdim + j;
        float base = (ch == 0) ? (x1[idx] + bj) : out[idx];
        out[idx] = base + acc[mt][nt][r];
      }
    }
}

// ---------------- launch ----------------
extern "C" void kernel_launch(void* const* d_in, const int* in_sizes, int n_in,
                              void* d_out, int out_size, void* d_ws, size_t ws_size,
                              hipStream_t stream) {
  const float* x      = (const float*)d_in[0];
  const float* n1s    = (const float*)d_in[1];
  const float* n1b    = (const float*)d_in[2];
  const float* qkv_w  = (const float*)d_in[3];
  const float* qkv_b  = (const float*)d_in[4];
  const float* proj_w = (const float*)d_in[5];
  const float* proj_b = (const float*)d_in[6];
  const float* n2s    = (const float*)d_in[7];
  const float* n2b    = (const float*)d_in[8];
  const float* fc1_w  = (const float*)d_in[9];
  const float* fc1_b  = (const float*)d_in[10];
  const float* fc2_w  = (const float*)d_in[11];
  const float* fc2_b  = (const float*)d_in[12];
  float* out = (float*)d_out;
  (void)in_sizes; (void)n_in; (void)out_size; (void)ws_size;

  char* ws = (char*)d_ws;
  size_t off = 0;
  unsigned short* wsA  = (unsigned short*)(ws + off); off += (size_t)TOK * Cdim * 2;
  unsigned short* wsB  = (unsigned short*)(ws + off); off += (size_t)TOK * Cdim * 2;
  float*          wsX1 = (float*)(ws + off);          off += (size_t)TOK * Cdim * 4;
  unsigned short* qkvT = (unsigned short*)(ws + off); off += (size_t)1152 * 384 * 2;
  unsigned short* projT= (unsigned short*)(ws + off); off += (size_t)384 * 384 * 2;
  unsigned short* fc1T = (unsigned short*)(ws + off); off += (size_t)MLPH * 384 * 2;
  unsigned short* fc2T = (unsigned short*)(ws + off); off += (size_t)384 * MLPH * 2;

  // pack weights -> bf16 transposed
  pack_wT_kernel<<<(384 * 1152 + 255) / 256, 256, 0, stream>>>(qkv_w, qkvT, 384, 1152);
  pack_wT_kernel<<<(384 * 384 + 255) / 256, 256, 0, stream>>>(proj_w, projT, 384, 384);
  pack_wT_kernel<<<(384 * MLPH + 255) / 256, 256, 0, stream>>>(fc1_w, fc1T, 384, MLPH);
  pack_wT_kernel<<<(MLPH * 384 + 255) / 256, 256, 0, stream>>>(fc2_w, fc2T, MLPH, 384);

  // LN1 + shift + window partition
  ln_kernel<<<TOK / 8, 256, 0, stream>>>(x, n1s, n1b, wsA, 1);

  // windowed attention (QKV + softmax + PV), one wave per (window, head)
  attn_kernel<<<NWIN * NHd, 32, 0, stream>>>(wsA, qkvT, qkv_b, wsB);

  // proj + reverse shift + residual -> x1
  proj_kernel<<<MT64 * 6, 32, 0, stream>>>(wsB, projT, proj_b, x, wsX1);

  // LN2
  ln_kernel<<<TOK / 8, 256, 0, stream>>>(wsX1, n2s, n2b, wsA, 0);

  // MLP, chunked over MLPH in 4 slices of 384; fc2 ch0 seeds out with x1 residual
  for (int ch = 0; ch < 4; ++ch) {
    fc1_kernel<<<MT64 * 6, 32, 0, stream>>>(wsA, fc1T, fc1_b, wsB, ch);
    fc2_kernel<<<MT64 * 6, 32, 0, stream>>>(wsB, fc2T, fc2_b, wsX1, out, ch);
  }
}
